// WindowedSpatialAttention_89833535964097
// MI455X (gfx1250) — matrix operand order repaired and checked by reading, not compiled, not run on hardware
//
#include <hip/hip_runtime.h>

// ---------------------------------------------------------------------------
// Windowed spatial attention, fused per-window kernel for gfx1250 (CDNA5).
// One workgroup (256 threads = 8 wave32) per 8x8 window (N = 4096 windows).
// GEMMs via v_wmma_f32_16x16x32_bf16 (f32 accum); x staged into LDS with
// global_load_async_to_lds_b128 (ASYNCcnt); v stored transposed for the
// attn@v B operand.
// ---------------------------------------------------------------------------

#define B_DIM   2
#define T_DIM   8
#define C_DIM   384
#define HP_DIM  128
#define WP_DIM  128
#define WS      8
#define NHEAD   12
#define DH      32
#define QKV_DIM (3 * C_DIM)   // 1152
#define S_TOK   64            // tokens per window
#define SPATIAL (HP_DIM * WP_DIM)  // 16384

// LDS layout (dynamic shared, bytes):
//  region A [0, 151552):
//    stage 1 : x f32 channel-major [384][XCS=68]          (104448 B)
//    stage 2+: qk bf16 [64][QS=776] at 0                  ( 99328 B)
//              vT bf16 [384][VS=68] at 99328              ( 52224 B)
//  region B [151552, 225280):
//    xn bf16 [64][XS=392] (dead after QKV)                ( 50176 B)
//    attn staging bf16 per-wave [64][AS=72] (overlay)     ( 73728 B)
constexpr int QS  = 776;   // qk row stride (bf16), 388 dw % 64 = 4
constexpr int VS  = 68;    // vT row stride (bf16)
constexpr int XCS = 68;    // x f32 staging stride (floats), 68 dw % 64 = 4
constexpr int XS  = 392;   // xn row stride (bf16), 196 dw % 64 = 4
constexpr int AS  = 72;    // attn staging row stride (bf16)
constexpr int OFF_A    = 0;
constexpr int OFF_VT   = S_TOK * QS * 2;               // 99328
constexpr int SZ_A     = OFF_VT + C_DIM * VS * 2;      // 151552 (>= 104448 xstage)
constexpr int OFF_XN   = SZ_A;
constexpr int OFF_ATTN = SZ_A;
constexpr int ATTN_WAVE_BYTES = S_TOK * AS * 2;        // 9216
constexpr int SMEM_BYTES = SZ_A + 8 * ATTN_WAVE_BYTES; // 225280

typedef __attribute__((ext_vector_type(16))) __bf16    v16bf;
typedef __attribute__((ext_vector_type(8)))  float     v8f;
typedef __attribute__((ext_vector_type(8)))  unsigned  v8u;

__device__ __forceinline__ unsigned short f32_to_bf16(float f) {
    unsigned u = __builtin_bit_cast(unsigned, f);
    unsigned r = 0x7FFFu + ((u >> 16) & 1u);   // round to nearest even
    return (unsigned short)((u + r) >> 16);
}

__device__ __forceinline__ v8f wmma_bf16(v16bf a, v16bf b, v8f c) {
    return __builtin_amdgcn_wmma_f32_16x16x32_bf16(
        false, a, false, b, (short)0, c, false, false);
}

// A fragment (16x32 bf16): lanes 0-15 kh=0, lanes 16-31 kh=8.
// VGPR v<4: K = kh+2v; v>=4: K = 16+kh+2(v-4). Memory row-major, pairs along K.
__device__ __forceinline__ v16bf frag_A(const unsigned short* base, int stride,
                                        int m0, int k0, int lane) {
    const unsigned short* row = base + (m0 + (lane & 15)) * stride;
    int kh = (lane < 16) ? 0 : 8;
    v8u u;
#pragma unroll
    for (int v = 0; v < 4; ++v) {
        u[v]     = *(const unsigned*)(row + k0 + kh + 2 * v);
        u[v + 4] = *(const unsigned*)(row + k0 + 16 + kh + 2 * v);
    }
    return __builtin_bit_cast(v16bf, u);
}

// B fragment (32x16 bf16) where memory holds Bt[n][k] (row n, contiguous k),
// i.e. B[k][n] = Bt[n][k]. lanes 0-15 kh=0, lanes 16-31 kh=8; same K order as frag_A.
__device__ __forceinline__ v16bf frag_Bt(const unsigned short* base, int stride,
                                         int n0, int k0, int lane) {
    const unsigned short* row = base + (size_t)(n0 + (lane & 15)) * stride;
    int kh = (lane < 16) ? 0 : 8;
    v8u u;
#pragma unroll
    for (int v = 0; v < 4; ++v) {
        u[v]     = *(const unsigned*)(row + k0 + kh + 2 * v);
        u[v + 4] = *(const unsigned*)(row + k0 + 16 + kh + 2 * v);
    }
    return __builtin_bit_cast(v16bf, u);
}

// C/D fragment (16x16 f32) -> bf16 row-major store (strided rows).
__device__ __forceinline__ void store_c_bf16(unsigned short* dst, int stride,
                                             int m0, int n0, int lane, v8f acc) {
    int col = n0 + (lane & 15);
    int mb  = m0 + ((lane < 16) ? 0 : 8);
#pragma unroll
    for (int r = 0; r < 8; ++r)
        dst[(mb + r) * stride + col] = f32_to_bf16(acc[r]);
}

// Async copy: global (saddr + voff) -> LDS (lds_off), 16B per lane, ASYNCcnt.
__device__ __forceinline__ void async_ld_b128(unsigned lds_off, const float* saddr,
                                              unsigned voff) {
    asm volatile("global_load_async_to_lds_b128 %0, %1, %2"
                 :: "v"(lds_off), "v"(voff), "s"(saddr)
                 : "memory");
}

// ---------------------------------------------------------------------------
__global__ void cvt_weights_bf16(const float* __restrict__ wqkv,
                                 const float* __restrict__ wout,
                                 const float* __restrict__ wproj,
                                 unsigned short* __restrict__ dst) {
    const int NQ = QKV_DIM * C_DIM;   // 442368
    const int NO = C_DIM * C_DIM;     // 147456
    int i = blockIdx.x * 256 + threadIdx.x;
    if (i < NQ)               dst[i] = f32_to_bf16(wqkv[i]);
    else if (i < NQ + NO)     dst[i] = f32_to_bf16(wout[i - NQ]);
    else if (i < NQ + 2 * NO) dst[i] = f32_to_bf16(wproj[i - NQ - NO]);
}

// ---------------------------------------------------------------------------
__global__ void __launch_bounds__(256)
fused_window_attn(const float* __restrict__ x,
                  const float* __restrict__ gamma,
                  const float* __restrict__ beta,
                  const unsigned short* __restrict__ wqkv,
                  const float* __restrict__ bqkv,
                  const unsigned short* __restrict__ wout,
                  const float* __restrict__ bout,
                  const unsigned short* __restrict__ wproj,
                  const float* __restrict__ bproj,
                  float* __restrict__ out) {
    extern __shared__ char smem[];
    unsigned short* qk_lds = (unsigned short*)(smem + OFF_A);
    unsigned short* vt_lds = (unsigned short*)(smem + OFF_VT);
    unsigned short* xn_lds = (unsigned short*)(smem + OFF_XN);
    float*          xstage = (float*)(smem + OFF_A);   // stage-1 alias, [384][XCS]

    __shared__ float s_sum[256];
    __shared__ float s_sq[256];

    const int tid  = threadIdx.x;
    const int lane = tid & 31;
    const int wave = tid >> 5;

    const int wid = blockIdx.x;
    const int bt  = wid >> 8;
    const int wh  = (wid >> 4) & 15;
    const int ww  = wid & 15;
    const int sp0 = (wh * WS) * WP_DIM + ww * WS;

    // ---- stage 1a: async-copy x window into LDS, channel-major f32 ----
    {
        const float* xw = x + (size_t)bt * C_DIM * SPATIAL + sp0;
        const unsigned dynbase = __builtin_amdgcn_groupstaticsize();
#pragma unroll
        for (int t = 0; t < 24; ++t) {                  // 6144 quads / 256 thr
            int idx  = tid + t * 256;
            int c    = idx >> 4;
            int tok0 = (idx & 15) * 4;                  // 4 tokens per b128
            int i = tok0 >> 3, j = tok0 & 7;            // j in {0,4}
            unsigned voff = (unsigned)((c * SPATIAL + i * WP_DIM + j) * 4);
            unsigned loff = dynbase + (unsigned)((c * XCS + tok0) * 4);
            async_ld_b128(loff, xw, voff);
        }
        asm volatile("s_wait_asynccnt 0" ::: "memory");
    }
    __syncthreads();

    // ---- stage 1b: LayerNorm over C, write xn bf16 row-major ----
    {
        const int tk = tid & 63, qr = tid >> 6;
        const int c0 = qr * 96;
        float sum = 0.f, sq = 0.f;
        for (int c = c0; c < c0 + 96; ++c) {
            float v = xstage[c * XCS + tk];
            sum += v; sq += v * v;
        }
        s_sum[tk * 4 + qr] = sum;
        s_sq [tk * 4 + qr] = sq;
        __syncthreads();
        float m  = (s_sum[tk*4] + s_sum[tk*4+1] + s_sum[tk*4+2] + s_sum[tk*4+3]) * (1.f / C_DIM);
        float m2 = (s_sq [tk*4] + s_sq [tk*4+1] + s_sq [tk*4+2] + s_sq [tk*4+3]) * (1.f / C_DIM);
        float rs = rsqrtf(m2 - m * m + 1e-5f);
        for (int c = c0; c < c0 + 96; ++c) {
            float v = (xstage[c * XCS + tk] - m) * rs;
            xn_lds[tk * XS + c] = f32_to_bf16(v * gamma[c] + beta[c]);
        }
    }
    __syncthreads();   // xstage dead; region A free for qk + vT

    // ---- stage 2: QKV = xn(64x384) @ Wqkv^T + b; v stored transposed ----
    for (int nt = wave; nt < QKV_DIM / 16; nt += 8) {
        int n0 = nt * 16;
        float bias = bqkv[n0 + (lane & 15)];
        for (int mt = 0; mt < 4; ++mt) {
            v8f acc;
#pragma unroll
            for (int r = 0; r < 8; ++r) acc[r] = bias;
            for (int kb = 0; kb < C_DIM; kb += 32) {
                v16bf a = frag_A(xn_lds, XS, mt * 16, kb, lane);
                v16bf b = frag_Bt(wqkv, C_DIM, n0, kb, lane);
                acc = wmma_bf16(a, b, acc);
            }
            if (n0 < 2 * C_DIM) {
                store_c_bf16(qk_lds, QS, mt * 16, n0, lane, acc);   // q,k rows
            } else {
                // v: per-lane C column = 8 consecutive tokens of one channel
                int dv = n0 - 2 * C_DIM + (lane & 15);
                int mb = mt * 16 + ((lane < 16) ? 0 : 8);
                unsigned d01 = f32_to_bf16(acc[0]) | ((unsigned)f32_to_bf16(acc[1]) << 16);
                unsigned d23 = f32_to_bf16(acc[2]) | ((unsigned)f32_to_bf16(acc[3]) << 16);
                unsigned d45 = f32_to_bf16(acc[4]) | ((unsigned)f32_to_bf16(acc[5]) << 16);
                unsigned d67 = f32_to_bf16(acc[6]) | ((unsigned)f32_to_bf16(acc[7]) << 16);
                unsigned* p = (unsigned*)(vt_lds + dv * VS + mb);   // 8B aligned
                p[0] = d01; p[1] = d23; p[2] = d45; p[3] = d67;
            }
        }
    }
    __syncthreads();

    // ---- stage 3: per-head attention (wave w -> heads w, w+8) ----
    unsigned short* attst = (unsigned short*)(smem + OFF_ATTN + wave * ATTN_WAVE_BYTES);
    const float scale = 0.17677669529663687f;  // 1/sqrt(32)
    for (int h = wave; h < NHEAD; h += 8) {
        const int qc = h * DH;
        const int kc = C_DIM + h * DH;

        for (int mt = 0; mt < 4; ++mt) {
            v8f sc[4];
            for (int nt = 0; nt < 4; ++nt) {
                v16bf a = frag_A(qk_lds, QS, mt * 16, qc, lane);    // q
                v16bf b = frag_Bt(qk_lds, QS, nt * 16, kc, lane);   // k^T
                v8f z;
#pragma unroll
                for (int r = 0; r < 8; ++r) z[r] = 0.f;
                sc[nt] = wmma_bf16(a, b, z);
            }
#pragma unroll
            for (int nt = 0; nt < 4; ++nt)
#pragma unroll
                for (int r = 0; r < 8; ++r) sc[nt][r] *= scale;

#pragma unroll
            for (int r = 0; r < 8; ++r) {
                float mx = -3.4e38f;
#pragma unroll
                for (int nt = 0; nt < 4; ++nt) mx = fmaxf(mx, sc[nt][r]);
                for (int msk = 1; msk <= 8; msk <<= 1) mx = fmaxf(mx, __shfl_xor(mx, msk));
                float sum = 0.f;
#pragma unroll
                for (int nt = 0; nt < 4; ++nt) {
                    float e = __expf(sc[nt][r] - mx);
                    sc[nt][r] = e;
                    sum += e;
                }
                for (int msk = 1; msk <= 8; msk <<= 1) sum += __shfl_xor(sum, msk);
                float inv = 1.f / sum;
                int row = mt * 16 + r + ((lane < 16) ? 0 : 8);
#pragma unroll
                for (int nt = 0; nt < 4; ++nt)
                    attst[row * AS + nt * 16 + (lane & 15)] = f32_to_bf16(sc[nt][r] * inv);
            }
        }

        // o = attn(64x64) @ v(64x32) ; vT rows = channels, contiguous tokens
        for (int mt = 0; mt < 4; ++mt) {
            for (int nj = 0; nj < 2; ++nj) {
                v8f acc;
#pragma unroll
                for (int r = 0; r < 8; ++r) acc[r] = 0.f;
                for (int kt = 0; kt < 2; ++kt) {
                    v16bf a = frag_A(attst, AS, mt * 16, kt * 32, lane);
                    v16bf b = frag_Bt(vt_lds, VS, h * DH + nj * 16, kt * 32, lane);
                    acc = wmma_bf16(a, b, acc);
                }
                store_c_bf16(qk_lds, QS, mt * 16, qc + nj * 16, lane, acc);  // o -> q slot
            }
        }
    }
    __syncthreads();

    // ---- stage 4: out_proj: r1 = o(64x384) @ Wout^T + b -> k slot ----
    for (int nt = wave; nt < C_DIM / 16; nt += 8) {
        int n0 = nt * 16;
        float bias = bout[n0 + (lane & 15)];
        for (int mt = 0; mt < 4; ++mt) {
            v8f acc;
#pragma unroll
            for (int r = 0; r < 8; ++r) acc[r] = bias;
            for (int kb = 0; kb < C_DIM; kb += 32) {
                v16bf a = frag_A(qk_lds, QS, mt * 16, kb, lane);        // o cols [0,384)
                v16bf b = frag_Bt(wout, C_DIM, n0, kb, lane);
                acc = wmma_bf16(a, b, acc);
            }
            store_c_bf16(qk_lds, QS, mt * 16, C_DIM + n0, lane, acc);   // -> cols [384,768)
        }
    }
    __syncthreads();

    // ---- stage 5: proj: r2 = r1 @ Wproj^T + b -> scatter f32 to global ----
    float* ob = out + (size_t)bt * C_DIM * SPATIAL + sp0;
    for (int nt = wave; nt < C_DIM / 16; nt += 8) {
        int n0 = nt * 16;
        float bias = bproj[n0 + (lane & 15)];
        for (int mt = 0; mt < 4; ++mt) {
            v8f acc;
#pragma unroll
            for (int r = 0; r < 8; ++r) acc[r] = bias;
            for (int kb = 0; kb < C_DIM; kb += 32) {
                v16bf a = frag_A(qk_lds, QS, mt * 16, C_DIM + kb, lane);
                v16bf b = frag_Bt(wproj, C_DIM, n0, kb, lane);
                acc = wmma_bf16(a, b, acc);
            }
            int c  = n0 + (lane & 15);
            int mb = mt * 16 + ((lane < 16) ? 0 : 8);
            float* och = ob + (size_t)c * SPATIAL;
#pragma unroll
            for (int r = 0; r < 8; ++r) {
                int m = mb + r;
                och[(m >> 3) * WP_DIM + (m & 7)] = acc[r];
            }
        }
    }
}

// ---------------------------------------------------------------------------
extern "C" void kernel_launch(void* const* d_in, const int* in_sizes, int n_in,
                              void* d_out, int out_size, void* d_ws, size_t ws_size,
                              hipStream_t stream) {
    const float* x     = (const float*)d_in[0];
    const float* gamma = (const float*)d_in[1];
    const float* beta  = (const float*)d_in[2];
    const float* wqkv  = (const float*)d_in[3];
    const float* bqkv  = (const float*)d_in[4];
    const float* wout  = (const float*)d_in[5];
    const float* bout  = (const float*)d_in[6];
    const float* wproj = (const float*)d_in[7];
    const float* bproj = (const float*)d_in[8];
    float* out = (float*)d_out;

    unsigned short* wbf = (unsigned short*)d_ws;
    const int NQ = QKV_DIM * C_DIM;   // 442368
    const int NO = C_DIM * C_DIM;     // 147456

    cvt_weights_bf16<<<(NQ + 2 * NO + 255) / 256, 256, 0, stream>>>(wqkv, wout, wproj, wbf);

    (void)hipFuncSetAttribute((const void*)fused_window_attn,
                              hipFuncAttributeMaxDynamicSharedMemorySize, SMEM_BYTES);

    const int n_windows = B_DIM * T_DIM * (HP_DIM / WS) * (WP_DIM / WS);  // 4096
    fused_window_attn<<<n_windows, 256, SMEM_BYTES, stream>>>(
        x, gamma, beta,
        wbf, bqkv,
        wbf + NQ, bout,
        wbf + NQ + NO, bproj,
        out);
}
